// CausalMultiHeadAttentionWithRoPE_34746285425443
// MI455X (gfx1250) — matrix operand order repaired and checked by reading, not compiled, not run on hardware
//
#include <hip/hip_runtime.h>
#include <hip/hip_bf16.h>

// ---------------------------------------------------------------------------
// Causal MHA with RoPE for MI455X (gfx1250): bf16 WMMA everywhere.
// B=2, S=2048, D=1024, H=16, d=64.
// ---------------------------------------------------------------------------

#define B_  2
#define S_  2048
#define D_  1024
#define H_  16
#define HD_ 64
#define M_  (B_ * S_)   // 4096 rows

typedef __attribute__((ext_vector_type(16))) __bf16 v16bf;
typedef __attribute__((ext_vector_type(8)))  __bf16 v8bf;
typedef __attribute__((ext_vector_type(8)))  float  v8f;

// ---------------------------------------------------------------------------
// fp32 -> bf16 elementwise convert
// ---------------------------------------------------------------------------
__global__ void f2bf_kernel(const float* __restrict__ src, __bf16* __restrict__ dst, int n) {
    int i = blockIdx.x * 256 + threadIdx.x;
    if (i < n) dst[i] = (__bf16)src[i];
}

// ---------------------------------------------------------------------------
// C[M,N] (f32) = A[M,K] (bf16, row major) @ W[N,K]^T (bf16, row major)
//
// Register-blocked: each wave computes a 32x64 tile of C as 2x4 WMMA tiles,
// reusing 2 A-fragments and 4 B-fragments per 32-wide K step
// (8 wmma per 12 b128 loads instead of 1 wmma per 4 loads).
//
// A-fragment layout (16-bit A 16x32): lane holds row M = lane%16;
//   half = lane/16 selects contiguous K-values {half*16..+15} (same order as B).
// B-fragment: lane holds column N = lane%16 of B (= row of W), contiguous
//   16 K-values starting at kk + half*16  -> 32B contiguous load of W's row.
// C/D layout: VGPR j, lanes0-15 -> M=j ; lanes16-31 -> M=8+j ; col = lane%16.
// ---------------------------------------------------------------------------
#define GEMM_MT 2   // 16-row tiles per wave
#define GEMM_NT 4   // 16-col tiles per wave

__global__ __launch_bounds__(256) void gemm_bf16_kernel(
    const __bf16* __restrict__ A, const __bf16* __restrict__ W,
    float* __restrict__ C, int Mdim, int Ndim, int Kdim) {
    const int lane = threadIdx.x & 31;
    const int wave = threadIdx.x >> 5;
    const int l16  = lane & 15;
    const int half = lane >> 4;

    const int tiles_n = Ndim / (16 * GEMM_NT);
    const int t  = blockIdx.x * (blockDim.x >> 5) + wave;
    const int mt = t / tiles_n;
    const int nt = t % tiles_n;
    const int mbase = mt * 16 * GEMM_MT;
    const int nbase = nt * 16 * GEMM_NT;
    if (mbase >= Mdim) return;

    const __bf16* arow[GEMM_MT];
    const __bf16* brow[GEMM_NT];
#pragma unroll
    for (int i = 0; i < GEMM_MT; ++i)
        arow[i] = A + (size_t)(mbase + i * 16 + l16) * Kdim + half * 16;
#pragma unroll
    for (int j = 0; j < GEMM_NT; ++j)
        brow[j] = W + (size_t)(nbase + j * 16 + l16) * Kdim + half * 16;

    v8f acc[GEMM_MT][GEMM_NT] = {};

    for (int kk = 0; kk < Kdim; kk += 32) {
        v16bf af[GEMM_MT], bfv[GEMM_NT];
#pragma unroll
        for (int i = 0; i < GEMM_MT; ++i) {
            v8bf a0 = *(const v8bf*)(arow[i] + kk);
            v8bf a1 = *(const v8bf*)(arow[i] + kk + 8);
#pragma unroll
            for (int e = 0; e < 8; ++e) { af[i][e] = a0[e]; af[i][e + 8] = a1[e]; }
        }
#pragma unroll
        for (int j = 0; j < GEMM_NT; ++j)
            bfv[j] = *(const v16bf*)(brow[j] + kk);

#pragma unroll
        for (int i = 0; i < GEMM_MT; ++i)
#pragma unroll
            for (int j = 0; j < GEMM_NT; ++j)
                acc[i][j] = __builtin_amdgcn_wmma_f32_16x16x32_bf16(
                    false, af[i], false, bfv[j], (short)0, acc[i][j], false, false);
    }

#pragma unroll
    for (int i = 0; i < GEMM_MT; ++i) {
#pragma unroll
        for (int j = 0; j < GEMM_NT; ++j) {
            float* crow = C + (size_t)(mbase + i * 16 + half * 8) * Ndim
                            + nbase + j * 16 + l16;
#pragma unroll
            for (int r = 0; r < 8; ++r) crow[(size_t)r * Ndim] = acc[i][j][r];
        }
    }
}

// ---------------------------------------------------------------------------
// RoPE + (B,S,H,d) -> (B,H,S,d) repack, fp32 -> bf16. apply=0 for V (copy).
// One thread per (even,odd) pair.
// ---------------------------------------------------------------------------
__global__ void rope_reshape_kernel(const float* __restrict__ src,
                                    const int* __restrict__ pos,
                                    __bf16* __restrict__ dst, int apply) {
    const int total = M_ * D_ / 2;
    int i = blockIdx.x * 256 + threadIdx.x;
    if (i >= total) return;
    int pair = i & (HD_ / 2 - 1);          // 0..31
    int h    = (i / (HD_ / 2)) & (H_ - 1); // 0..15
    int s    = (i / (D_ / 2)) & (S_ - 1);  // 0..2047
    int b    = i / (S_ * D_ / 2);

    const float* p = src + ((size_t)(b * S_ + s) * D_ + h * HD_ + pair * 2);
    float xe = p[0], xo = p[1];
    float o0 = xe, o1 = xo;
    if (apply) {
        float position = (float)pos[s];
        // inv_freq = theta^(-2*pair/d) ; log(10000) = 9.210340372f
        float inv = __expf(-(float)(2 * pair) * (9.210340371976184f / (float)HD_));
        float f = position * inv;
        float sn, cs;
        __sincosf(f, &sn, &cs);
        o0 = xe * cs - xo * sn;
        o1 = xe * sn + xo * cs;
    }
    __bf16* q = dst + ((size_t)((b * H_ + h) * S_ + s) * HD_ + pair * 2);
    q[0] = (__bf16)o0;
    q[1] = (__bf16)o1;
}

// ---------------------------------------------------------------------------
// Flash-style causal attention. Q,K,V: (B,H,S,d) bf16. O: (B*S, D) bf16.
// Block = 128 threads (4 waves); each wave owns 16 query rows; key blocks of
// 32. All waves run identical trip counts (masked tiles decay to 0).
// ---------------------------------------------------------------------------
__global__ __launch_bounds__(128) void attn_kernel(
    const __bf16* __restrict__ Q, const __bf16* __restrict__ K,
    const __bf16* __restrict__ V, __bf16* __restrict__ O) {
    __shared__ __bf16 Vt[64][40];        // transposed V tile: [dd][key], pad->16B-aligned rows
    __shared__ __bf16 Plds[4][16][32];   // per-wave probability staging

    const int lane = threadIdx.x & 31;
    const int wave = threadIdx.x >> 5;
    const int l16  = lane & 15;
    const int half = lane >> 4;

    const int qt = blockIdx.x;            // query tile (64 rows)
    const int bh = blockIdx.y;            // b*H + h
    const int b  = bh / H_;
    const int h  = bh % H_;

    const __bf16* Qh = Q + (size_t)bh * S_ * HD_;
    const __bf16* Kh = K + (size_t)bh * S_ * HD_;
    const __bf16* Vh = V + (size_t)bh * S_ * HD_;

    const int qbase = qt * 64 + wave * 16;

    // Hoisted Q fragments: d-chunks [0,32) and [32,64)
    v16bf af0, af1;
    {
        const __bf16* qr = Qh + (size_t)(qbase + l16) * HD_ + half * 16;
        v8bf a0 = *(const v8bf*)(qr);
        v8bf a1 = *(const v8bf*)(qr + 8);
        v8bf a2 = *(const v8bf*)(qr + 32);
        v8bf a3 = *(const v8bf*)(qr + 40);
#pragma unroll
        for (int e = 0; e < 8; ++e) {
            af0[e] = a0[e]; af0[e + 8] = a1[e];
            af1[e] = a2[e]; af1[e + 8] = a3[e];
        }
    }

    float mrow[8], srow[8];
    v8f accs[4] = {};
#pragma unroll
    for (int j = 0; j < 8; ++j) { mrow[j] = -3e38f; srow[j] = 0.0f; }

    const int kmax = qt * 64 + 63;
    const float sc = 0.125f;  // 1/sqrt(64)

    for (int k0 = 0; k0 <= kmax; k0 += 32) {
        // ---- cooperative V staging (transposed) ----
        __syncthreads();
        {
            int tt  = threadIdx.x;
            int key = tt >> 2;            // 0..31
            int dd0 = (tt & 3) * 16;
            const __bf16* vp = Vh + (size_t)(k0 + key) * HD_ + dd0;
#pragma unroll
            for (int e = 0; e < 16; ++e) Vt[dd0 + e][key] = vp[e];
        }
        __syncthreads();

        // ---- scores: two 16-key tiles, contraction over d=64 ----
        v8f c0 = {}, c1 = {};
        {
            const __bf16* kr0 = Kh + (size_t)(k0 + l16) * HD_ + half * 16;
            const __bf16* kr1 = Kh + (size_t)(k0 + 16 + l16) * HD_ + half * 16;
            v16bf kf;
            kf = *(const v16bf*)(kr0);
            c0 = __builtin_amdgcn_wmma_f32_16x16x32_bf16(false, af0, false, kf, (short)0, c0, false, false);
            kf = *(const v16bf*)(kr0 + 32);
            c0 = __builtin_amdgcn_wmma_f32_16x16x32_bf16(false, af1, false, kf, (short)0, c0, false, false);
            kf = *(const v16bf*)(kr1);
            c1 = __builtin_amdgcn_wmma_f32_16x16x32_bf16(false, af0, false, kf, (short)0, c1, false, false);
            kf = *(const v16bf*)(kr1 + 32);
            c1 = __builtin_amdgcn_wmma_f32_16x16x32_bf16(false, af1, false, kf, (short)0, c1, false, false);
        }

        // ---- scale + causal mask ----
        const int col0 = k0 + l16;
        const int col1 = col0 + 16;
#pragma unroll
        for (int j = 0; j < 8; ++j) {
            const int row = qbase + j + half * 8;
            float s0 = c0[j] * sc; if (col0 > row) s0 = -1e9f;
            float s1 = c1[j] * sc; if (col1 > row) s1 = -1e9f;
            c0[j] = s0; c1[j] = s1;
        }

        // ---- online softmax update (row reductions across 16 lanes) ----
#pragma unroll
        for (int j = 0; j < 8; ++j) {
            float mx = fmaxf(c0[j], c1[j]);
#pragma unroll
            for (int d = 1; d < 16; d <<= 1) mx = fmaxf(mx, __shfl_xor(mx, d, 32));
            float mnew = fmaxf(mrow[j], mx);
            float fac  = __expf(mrow[j] - mnew);
            float p0   = __expf(c0[j] - mnew);
            float p1   = __expf(c1[j] - mnew);
            float rs   = p0 + p1;
#pragma unroll
            for (int d = 1; d < 16; d <<= 1) rs += __shfl_xor(rs, d, 32);
            srow[j] = srow[j] * fac + rs;
            mrow[j] = mnew;
#pragma unroll
            for (int t = 0; t < 4; ++t) accs[t][j] *= fac;
            c0[j] = p0; c1[j] = p1;
        }

        // ---- stage P (C-layout) -> LDS, re-read as A fragment ----
#pragma unroll
        for (int j = 0; j < 8; ++j) {
            Plds[wave][j + half * 8][l16]      = (__bf16)c0[j];
            Plds[wave][j + half * 8][16 + l16] = (__bf16)c1[j];
        }
        asm volatile("s_wait_dscnt 0" ::: "memory");
        __builtin_amdgcn_wave_barrier();

        v16bf pf;
        {
            const __bf16* pr = &Plds[wave][l16][half * 16];
            v8bf p0v = *(const v8bf*)(pr);
            v8bf p1v = *(const v8bf*)(pr + 8);
#pragma unroll
            for (int e = 0; e < 8; ++e) { pf[e] = p0v[e]; pf[e + 8] = p1v[e]; }
        }

        // ---- PV: 4 output N-tiles of 16 ----
#pragma unroll
        for (int t = 0; t < 4; ++t) {
            v16bf vf = *(const v16bf*)(&Vt[t * 16 + l16][half * 16]);
            accs[t] = __builtin_amdgcn_wmma_f32_16x16x32_bf16(
                false, pf, false, vf, (short)0, accs[t], false, false);
        }
    }

    // ---- normalize + write (B*S, D) bf16 ----
#pragma unroll
    for (int t = 0; t < 4; ++t) {
#pragma unroll
        for (int j = 0; j < 8; ++j) {
            const int row = qbase + j + half * 8;
            float val = accs[t][j] / srow[j];
            O[(size_t)(b * S_ + row) * D_ + h * HD_ + t * 16 + l16] = (__bf16)val;
        }
    }
}

// ---------------------------------------------------------------------------
// Launch
// ---------------------------------------------------------------------------
extern "C" void kernel_launch(void* const* d_in, const int* in_sizes, int n_in,
                              void* d_out, int out_size, void* d_ws, size_t ws_size,
                              hipStream_t stream) {
    const float* x  = (const float*)d_in[0];
    const float* wq = (const float*)d_in[1];
    const float* wk = (const float*)d_in[2];
    const float* wv = (const float*)d_in[3];
    const float* wo = (const float*)d_in[4];
    const int*   pos = (const int*)d_in[5];
    float* out = (float*)d_out;

    // workspace carve-up
    char* ws = (char*)d_ws;
    __bf16* xb  = (__bf16*)ws;                          // M*D
    __bf16* wqb = xb  + (size_t)M_ * D_;                // D*D
    __bf16* wkb = wqb + (size_t)D_ * D_;
    __bf16* wvb = wkb + (size_t)D_ * D_;
    __bf16* wob = wvb + (size_t)D_ * D_;
    float*  tmpf = (float*)(wob + (size_t)D_ * D_);     // M*D f32 (reused Q/K/V)
    __bf16* Qb  = (__bf16*)(tmpf + (size_t)M_ * D_);    // (B,H,S,d)
    __bf16* Kb  = Qb + (size_t)M_ * D_;
    __bf16* Vb  = Kb + (size_t)M_ * D_;
    __bf16* Ab  = Vb + (size_t)M_ * D_;                 // attention out (B*S, D)

    const int nAct = M_ * D_;
    const int nW   = D_ * D_;
    f2bf_kernel<<<(nAct + 255) / 256, 256, 0, stream>>>(x,  xb,  nAct);
    f2bf_kernel<<<(nW   + 255) / 256, 256, 0, stream>>>(wq, wqb, nW);
    f2bf_kernel<<<(nW   + 255) / 256, 256, 0, stream>>>(wk, wkb, nW);
    f2bf_kernel<<<(nW   + 255) / 256, 256, 0, stream>>>(wv, wvb, nW);
    f2bf_kernel<<<(nW   + 255) / 256, 256, 0, stream>>>(wo, wob, nW);

    // 32x64 tile per wave, 8 waves per block
    const int waveTiles  = (M_ / (16 * GEMM_MT)) * (D_ / (16 * GEMM_NT));
    const int gemmBlocks = waveTiles / 8;
    const int ropeBlocks = (M_ * D_ / 2 + 255) / 256;

    gemm_bf16_kernel<<<gemmBlocks, 256, 0, stream>>>(xb, wqb, tmpf, M_, D_, D_);
    rope_reshape_kernel<<<ropeBlocks, 256, 0, stream>>>(tmpf, pos, Qb, 1);

    gemm_bf16_kernel<<<gemmBlocks, 256, 0, stream>>>(xb, wkb, tmpf, M_, D_, D_);
    rope_reshape_kernel<<<ropeBlocks, 256, 0, stream>>>(tmpf, pos, Kb, 1);

    gemm_bf16_kernel<<<gemmBlocks, 256, 0, stream>>>(xb, wvb, tmpf, M_, D_, D_);
    rope_reshape_kernel<<<ropeBlocks, 256, 0, stream>>>(tmpf, pos, Vb, 0);

    dim3 gAttn(S_ / 64, B_ * H_);
    attn_kernel<<<gAttn, 128, 0, stream>>>(Qb, Kb, Vb, Ab);

    gemm_bf16_kernel<<<gemmBlocks, 256, 0, stream>>>(Ab, wob, out, M_, D_, D_);
}
